// MP_PDE_Solver_46488726012309
// MI455X (gfx1250) — matrix operand order repaired and checked by reading, not compiled, not run on hardware
//
#include <hip/hip_runtime.h>
#include <hip/hip_bf16.h>
#include <stdint.h>

// ---------------- problem constants ----------------
#define N_NODES 16384
#define N_EDGES 262144
#define TW_     25
#define HDIM    128
#define NB_     8
#define NLAY    6
#define MSG_KPAD 288   // 284 padded to 9*32
#define UPD_KPAD 288   // 257 padded to 9*32
#define EMB_KPAD 32    // 28 padded to 32

typedef __attribute__((ext_vector_type(16))) __bf16       v16bf;
typedef __attribute__((ext_vector_type(8)))  float        v8f;
typedef __attribute__((ext_vector_type(4)))  unsigned int v4u;
typedef __attribute__((ext_vector_type(8)))  int          v8i;
typedef __attribute__((ext_vector_type(4)))  int          v4i;

union FragU { v16bf v; unsigned short s[16]; };

__device__ __forceinline__ unsigned short f2bf(float f) {
  unsigned b = __float_as_uint(f);
  return (unsigned short)((b + 0x7FFFu + ((b >> 16) & 1u)) >> 16);
}

// Intra-wave LDS producer->consumer sync (cross-lane): wait all DS ops done.
__device__ __forceinline__ void wave_lds_sync() {
  asm volatile("s_wait_dscnt 0x0" ::: "memory");
}

__device__ __forceinline__ v8f wmma_bf16(v16bf a, v16bf b, v8f c) {
#if defined(__HIP_DEVICE_COMPILE__)
  // D(16x16 f32) = A(16x32 bf16) x B(32x16 bf16) + C
  return __builtin_amdgcn_wmma_f32_16x16x32_bf16(false, a, false, b, (short)0, c,
                                                 false, false);
#else
  (void)a; (void)b;
  return c;
#endif
}

// A fragment: rows = lane&15, lanes<16 hold K {0..7,16..23}, lanes>=16 {8..15,24..31}
__device__ __forceinline__ v16bf load_a_frag(const unsigned short* A, int stride,
                                             int lane, int kc) {
  const int row = lane & 15, hi = lane >> 4;
  const unsigned short* p = A + row * stride + kc * 32 + hi * 8;
  FragU f;
#pragma unroll
  for (int e = 0; e < 8; ++e)  f.s[e] = p[e];        // K = kc*32 + hi*8 + e
#pragma unroll
  for (int e = 8; e < 16; ++e) f.s[e] = p[e + 8];    // K = kc*32 + hi*8 + e + 8
  return f.v;
}

// B fragment: pre-packed so each lane reads 16 contiguous bf16 (2x ds_load_b128).
__device__ __forceinline__ v16bf load_b_frag(const unsigned short* Wf, int lane,
                                             int kc, int nt) {
  const unsigned short* p = Wf + ((((kc << 3) + nt) << 5) + lane) * 16;
  FragU f;
#pragma unroll
  for (int e = 0; e < 16; ++e) f.s[e] = p[e];
  return f.v;
}

// ---------------- TDM weight staging ----------------
#if defined(__HIP_DEVICE_COMPILE__) && __has_builtin(__builtin_amdgcn_tensor_load_to_lds)
#define USE_TDM 1
__device__ __forceinline__ void tdm_copy_to_lds(const void* gsrc, unsigned ldsOff,
                                                unsigned nbytes) {
  unsigned long long ga = (unsigned long long)(uintptr_t)gsrc;
  unsigned n8 = nbytes >> 3;  // 8-byte elements, contiguous 1-D tile
  v4u g0;
  g0[0] = 1u;                                                  // count=1, user D#
  g0[1] = ldsOff;                                              // lds_addr
  g0[2] = (unsigned)ga;                                        // global_addr[31:0]
  g0[3] = (unsigned)((ga >> 32) & 0x1FFFFFFull) | (2u << 30);  // addr[56:32] | type=2
  v8i g1;
  g1[0] = (int)(3u << 16);                              // data_size = 8B
  g1[1] = (int)((n8 & 0xFFFFu) << 16);                  // tensor_dim0[15:0]
  g1[2] = (int)(((n8 >> 16) & 0xFFFFu) | (1u << 16));   // tensor_dim0[31:16], dim1=1
  g1[3] = (int)((n8 & 0xFFFFu) << 16);                  // tile_dim0 = n8
  g1[4] = 0;                                            // tile_dim1/2 unused
  g1[5] = (int)n8;                                      // tensor_dim0_stride
  g1[6] = 0;
  g1[7] = 0;
  v4i z4 = {0, 0, 0, 0};                                // groups 2/3 disabled (2-D)
  v8i z8 = {0, 0, 0, 0, 0, 0, 0, 0};
  // clang-23 / therock 6-arg form: (g0, g1, g2, g3, g_extra, cpol)
  __builtin_amdgcn_tensor_load_to_lds(g0, g1, z4, z4, z8, 0);
}
#endif

__device__ __forceinline__ void stage_weights(const unsigned short* g1, unsigned short* l1,
                                              int n1, const unsigned short* g2,
                                              unsigned short* l2, int n2, int tid) {
#if defined(USE_TDM)
  if (tid < 32) {  // one wave issues the DMA; TDM ignores EXEC
    tdm_copy_to_lds(g1, (unsigned)(uintptr_t)l1, (unsigned)(n1 * 2));
    tdm_copy_to_lds(g2, (unsigned)(uintptr_t)l2, (unsigned)(n2 * 2));
    __builtin_amdgcn_s_wait_tensorcnt(0);
  }
#else
  for (int i = tid; i < n1; i += 256) l1[i] = g1[i];
  for (int i = tid; i < n2; i += 256) l2[i] = g2[i];
#endif
}

// ---------------- small utility kernels ----------------
__global__ void zero_kernel(float* __restrict__ p, int n) {
  int t = blockIdx.x * 256 + threadIdx.x;
  if (t < n) p[t] = 0.f;
}

__global__ void count_deg_kernel(const int* __restrict__ ei, float* __restrict__ deg) {
  int t = blockIdx.x * 256 + threadIdx.x;
  if (t < N_EDGES) atomicAdd(&deg[ei[N_EDGES + t]], 1.f);
}

__global__ void count_batch_kernel(const int* __restrict__ b, float* __restrict__ c) {
  int t = blockIdx.x * 256 + threadIdx.x;
  if (t < N_NODES) atomicAdd(&c[b[t]], 1.f);
}

// Repack f32 weights [Kin x 128] into bf16 WMMA-B fragment order, K padded to Kpad.
// dst[((kc*8+nt)*32 + lane)*16 + e] = W[k][n], k = kc*32 + (lane>>4)*8 + (e&7) + (e>>3)*16, n = nt*16 + (lane&15)
__global__ void pack_weights_kernel(const float* __restrict__ W, int Kin, int Kpad,
                                    unsigned short* __restrict__ dst) {
  int t = blockIdx.x * 256 + threadIdx.x;
  if (t >= Kpad * 128) return;
  int e = t & 15;
  int lane = (t >> 4) & 31;
  int blk = t >> 9;
  int kc = blk >> 3, nt = blk & 7;
  int k = kc * 32 + (lane >> 4) * 8 + (e & 7) + (e >> 3) * 16;
  int n = nt * 16 + (lane & 15);
  float v = (k < Kin) ? W[k * 128 + n] : 0.f;
  dst[t] = f2bf(v);
}

// ---------------- embedding MLP: 28 -> 128 -> 128 ----------------
__global__ void __launch_bounds__(256) embed_kernel(
    const float* __restrict__ u, const float* __restrict__ pos,
    const unsigned short* __restrict__ W1f, const float* __restrict__ b1,
    const unsigned short* __restrict__ W2f, const float* __restrict__ b2,
    float* __restrict__ h, unsigned short* __restrict__ hbf) {
  extern __shared__ unsigned short smem[];
  unsigned short* W1s = smem;                    // 1*8*32*16 = 4096
  unsigned short* W2s = W1s + 1 * 8 * 32 * 16;   // 4*8*32*16 = 16384
  unsigned short* Abase = W2s + 4 * 8 * 32 * 16; // 8*16*32
  unsigned short* HidBase = Abase + 8 * 16 * EMB_KPAD;

  const int tid = threadIdx.x, wave = tid >> 5, lane = tid & 31;
  stage_weights(W1f, W1s, 1 * 8 * 32 * 16, W2f, W2s, 4 * 8 * 32 * 16, tid);
  __syncthreads();

  unsigned short* A = Abase + wave * (16 * EMB_KPAD);
  unsigned short* Hid = HidBase + wave * (16 * HDIM);
  const int row = lane & 15, hi = lane >> 4;
  const int nodeBase = (blockIdx.x * 8 + wave) * 16;
  const int node = nodeBase + row;

  for (int f = hi * 16; f < hi * 16 + 16; ++f) {
    float x;
    if (f < TW_)       x = u[node * TW_ + f];
    else if (f == 25)  x = pos[node * 3 + 1];   // pos_sp x
    else if (f == 26)  x = pos[node * 3 + 2];   // pos_sp y
    else if (f == 27)  x = pos[node * 3 + 0];   // variables
    else               x = 0.f;
    A[row * EMB_KPAD + f] = f2bf(x);
  }
  wave_lds_sync();

  for (int nt = 0; nt < 8; ++nt) {  // hidden = relu(in @ W1 + b1)
    v8f c = {};
    c = wmma_bf16(load_a_frag(A, EMB_KPAD, lane, 0), load_b_frag(W1s, lane, 0, nt), c);
    const float bias = b1[(nt << 4) + row];
#pragma unroll
    for (int r = 0; r < 8; ++r) {
      float vv = c[r] + bias; vv = vv > 0.f ? vv : 0.f;
      Hid[(r + hi * 8) * HDIM + (nt << 4) + row] = f2bf(vv);
    }
  }
  wave_lds_sync();

  for (int nt = 0; nt < 8; ++nt) {  // h = relu(hidden @ W2 + b2)
    v8f c = {};
#pragma unroll
    for (int kc = 0; kc < 4; ++kc)
      c = wmma_bf16(load_a_frag(Hid, HDIM, lane, kc), load_b_frag(W2s, lane, kc, nt), c);
    const float bias = b2[(nt << 4) + row];
#pragma unroll
    for (int r = 0; r < 8; ++r) {
      float vv = c[r] + bias; vv = vv > 0.f ? vv : 0.f;
      int n2 = nodeBase + r + hi * 8;
      h[n2 * HDIM + (nt << 4) + row] = vv;
      hbf[n2 * HDIM + (nt << 4) + row] = f2bf(vv);
    }
  }
}

// ---------------- edge message MLP: 284 -> 128 -> 128, scatter-add ----------------
__global__ void __launch_bounds__(256) message_kernel(
    const unsigned short* __restrict__ hbf, const float* __restrict__ u,
    const float* __restrict__ pos, const int* __restrict__ ei,
    const unsigned short* __restrict__ W1f, const float* __restrict__ b1,
    const unsigned short* __restrict__ W2f, const float* __restrict__ b2,
    float* __restrict__ agg) {
  extern __shared__ unsigned short smem[];
  unsigned short* W1s = smem;                               // 9*8*32*16 = 36864
  unsigned short* W2s = W1s + 9 * 8 * 32 * 16;              // 4*8*32*16 = 16384
  unsigned short* Abase = W2s + 4 * 8 * 32 * 16;            // 8*16*288
  int* idxBase = (int*)(Abase + 8 * 16 * MSG_KPAD);         // 8*16 ints
  unsigned short* HidBase = (unsigned short*)(idxBase + 8 * 16);

  const int tid = threadIdx.x, wave = tid >> 5, lane = tid & 31;
  stage_weights(W1f, W1s, 9 * 8 * 32 * 16, W2f, W2s, 4 * 8 * 32 * 16, tid);
  __syncthreads();

  unsigned short* A = Abase + wave * (16 * MSG_KPAD);
  unsigned short* Hid = HidBase + wave * (16 * HDIM);
  int* idxs = idxBase + wave * 16;

  const int row = lane & 15, hi = lane >> 4;
  const int e = (blockIdx.x * 8 + wave) * 16 + row;
  const int sN = ei[e];
  const int dN = ei[N_EDGES + e];
  if (hi == 0) idxs[row] = dN;

  // m_in = [h[dst], h[src], u[dst]-u[src], dpos, var[dst], pad]
  for (int f = hi * 144; f < hi * 144 + 144; ++f) {
    unsigned short v;
    if (f < 128)       v = hbf[dN * HDIM + f];
    else if (f < 256)  v = hbf[sN * HDIM + (f - 128)];
    else if (f < 281)  v = f2bf(u[dN * TW_ + (f - 256)] - u[sN * TW_ + (f - 256)]);
    else if (f == 281) v = f2bf(pos[dN * 3 + 1] - pos[sN * 3 + 1]);
    else if (f == 282) v = f2bf(pos[dN * 3 + 2] - pos[sN * 3 + 2]);
    else if (f == 283) v = f2bf(pos[dN * 3 + 0]);
    else               v = 0;
    A[row * MSG_KPAD + f] = v;
  }
  wave_lds_sync();

  for (int nt = 0; nt < 8; ++nt) {
    v8f c = {};
#pragma unroll
    for (int kc = 0; kc < 9; ++kc)
      c = wmma_bf16(load_a_frag(A, MSG_KPAD, lane, kc), load_b_frag(W1s, lane, kc, nt), c);
    const float bias = b1[(nt << 4) + row];
#pragma unroll
    for (int r = 0; r < 8; ++r) {
      float vv = c[r] + bias; vv = vv > 0.f ? vv : 0.f;
      Hid[(r + hi * 8) * HDIM + (nt << 4) + row] = f2bf(vv);
    }
  }
  wave_lds_sync();

  for (int nt = 0; nt < 8; ++nt) {
    v8f c = {};
#pragma unroll
    for (int kc = 0; kc < 4; ++kc)
      c = wmma_bf16(load_a_frag(Hid, HDIM, lane, kc), load_b_frag(W2s, lane, kc, nt), c);
    const float bias = b2[(nt << 4) + row];
#pragma unroll
    for (int r = 0; r < 8; ++r) {
      float vv = c[r] + bias; vv = vv > 0.f ? vv : 0.f;
      const int dd = idxs[r + hi * 8];
      atomicAdd(&agg[dd * HDIM + (nt << 4) + row], vv);  // segment-sum, L2-resident
    }
  }
}

// ---------------- node update MLP: 257 -> 128 -> 128 + norm stats ----------------
__global__ void __launch_bounds__(256) update_kernel(
    const unsigned short* __restrict__ hbf, const float* __restrict__ h,
    const float* __restrict__ agg, const float* __restrict__ deg,
    const float* __restrict__ pos, const int* __restrict__ batch,
    const unsigned short* __restrict__ W1f, const float* __restrict__ b1,
    const unsigned short* __restrict__ W2f, const float* __restrict__ b2,
    float* __restrict__ hnew, float* __restrict__ bsum, float* __restrict__ bsumsq) {
  extern __shared__ unsigned short smem[];
  unsigned short* W1s = smem;
  unsigned short* W2s = W1s + 9 * 8 * 32 * 16;
  unsigned short* Abase = W2s + 4 * 8 * 32 * 16;
  int* idxBase = (int*)(Abase + 8 * 16 * UPD_KPAD);
  unsigned short* HidBase = (unsigned short*)(idxBase + 8 * 16);

  const int tid = threadIdx.x, wave = tid >> 5, lane = tid & 31;
  stage_weights(W1f, W1s, 9 * 8 * 32 * 16, W2f, W2s, 4 * 8 * 32 * 16, tid);
  __syncthreads();

  unsigned short* A = Abase + wave * (16 * UPD_KPAD);
  unsigned short* Hid = HidBase + wave * (16 * HDIM);
  int* idxs = idxBase + wave * 16;

  const int row = lane & 15, hi = lane >> 4;
  const int nodeBase = (blockIdx.x * 8 + wave) * 16;
  const int node = nodeBase + row;
  if (hi == 0) idxs[row] = batch[node];

  float dg = deg[node];
  float invdeg = 1.f / (dg > 1.f ? dg : 1.f);

  // upd_in = [h, agg/deg, var, pad]
  for (int f = hi * 144; f < hi * 144 + 144; ++f) {
    unsigned short v;
    if (f < 128)       v = hbf[node * HDIM + f];
    else if (f < 256)  v = f2bf(agg[node * HDIM + (f - 128)] * invdeg);
    else if (f == 256) v = f2bf(pos[node * 3 + 0]);
    else               v = 0;
    A[row * UPD_KPAD + f] = v;
  }
  wave_lds_sync();

  for (int nt = 0; nt < 8; ++nt) {
    v8f c = {};
#pragma unroll
    for (int kc = 0; kc < 9; ++kc)
      c = wmma_bf16(load_a_frag(A, UPD_KPAD, lane, kc), load_b_frag(W1s, lane, kc, nt), c);
    const float bias = b1[(nt << 4) + row];
#pragma unroll
    for (int r = 0; r < 8; ++r) {
      float vv = c[r] + bias; vv = vv > 0.f ? vv : 0.f;
      Hid[(r + hi * 8) * HDIM + (nt << 4) + row] = f2bf(vv);
    }
  }
  wave_lds_sync();

  for (int nt = 0; nt < 8; ++nt) {
    v8f c = {};
#pragma unroll
    for (int kc = 0; kc < 4; ++kc)
      c = wmma_bf16(load_a_frag(Hid, HDIM, lane, kc), load_b_frag(W2s, lane, kc, nt), c);
    const float bias = b2[(nt << 4) + row];
#pragma unroll
    for (int r = 0; r < 8; ++r) {
      float vv = c[r] + bias; vv = vv > 0.f ? vv : 0.f;
      const int M = r + hi * 8;
      const int n2 = nodeBase + M;
      const int col = (nt << 4) + row;
      float hn = h[n2 * HDIM + col] + vv;  // residual
      hnew[n2 * HDIM + col] = hn;
      const int bb = idxs[M];
      atomicAdd(&bsum[bb * HDIM + col], hn);
      atomicAdd(&bsumsq[bb * HDIM + col], hn * hn);
    }
  }
}

__global__ void stats_kernel(const float* __restrict__ bsum, const float* __restrict__ bsumsq,
                             const float* __restrict__ bcnt, float* __restrict__ meanb,
                             float* __restrict__ rstdb) {
  int t = blockIdx.x * 256 + threadIdx.x;
  if (t >= NB_ * HDIM) return;
  int b = t >> 7;
  float cnt = fmaxf(bcnt[b], 1.f);
  float m = bsum[t] / cnt;
  float ms = bsumsq[t] / cnt;
  float var = fmaxf(ms - m * m, 0.f);
  meanb[t] = m;
  rstdb[t] = rsqrtf(var + 1e-5f);
}

__global__ void apply_norm_kernel(const float* __restrict__ hnew, const int* __restrict__ batch,
                                  const float* __restrict__ meanb, const float* __restrict__ rstdb,
                                  float* __restrict__ h, unsigned short* __restrict__ hbf) {
  int t = blockIdx.x * 256 + threadIdx.x;
  if (t >= N_NODES * HDIM) return;
  int node = t >> 7, f = t & 127;
  int b = batch[node];
  float v = (hnew[t] - meanb[b * HDIM + f]) * rstdb[b * HDIM + f];
  h[t] = v;
  hbf[t] = f2bf(v);
}

// ---------------- conv1d head, one wave per node ----------------
__global__ void __launch_bounds__(256) head_kernel(
    const float* __restrict__ h, const float* __restrict__ u,
    const float* __restrict__ c1W, const float* __restrict__ c1b,
    const float* __restrict__ c2W, const float* __restrict__ c2b,
    const float* __restrict__ c3W, const float* __restrict__ c3b,
    float* __restrict__ out) {
  __shared__ float sm[8][408];
  const int wave = threadIdx.x >> 5, lane = threadIdx.x & 31;
  const int node = blockIdx.x * 8 + wave;
  float* hrow = sm[wave];       // 128
  float* y1 = hrow + 128;       // 8*29 = 232
  float* y2 = y1 + 232;         // 4*10 = 40

  for (int i = lane; i < HDIM; i += 32) hrow[i] = h[node * HDIM + i];
  wave_lds_sync();

  for (int idx = lane; idx < 232; idx += 32) {  // conv1: k=15, stride 4
    int oc = idx / 29, t = idx % 29;
    float s = c1b[oc];
#pragma unroll
    for (int k = 0; k < 15; ++k) s += c1W[oc * 15 + k] * hrow[t * 4 + k];
    y1[idx] = s > 0.f ? s : 0.f;
  }
  wave_lds_sync();

  for (int idx = lane; idx < 40; idx += 32) {   // conv2: k=10, stride 2
    int oc = idx / 10, t = idx % 10;
    float s = c2b[oc];
    for (int ic = 0; ic < 8; ++ic)
#pragma unroll
      for (int k = 0; k < 10; ++k) s += c2W[(oc * 8 + ic) * 10 + k] * y1[ic * 29 + t * 2 + k];
    y2[idx] = s > 0.f ? s : 0.f;
  }
  wave_lds_sync();

  float diff = c3b[0];                          // conv3: k=10, stride 1 -> scalar
  for (int ic = 0; ic < 4; ++ic)
#pragma unroll
    for (int k = 0; k < 10; ++k) diff += c3W[ic * 10 + k] * y2[ic * 10 + k];

  if (lane < TW_) out[node * TW_ + lane] = u[node * TW_ + (TW_ - 1)] + 0.01f * (lane + 1) * diff;
}

// ---------------- workspace layout ----------------
static constexpr size_t OFF_H      = 0;
static constexpr size_t OFF_HNEW   = OFF_H      + (size_t)N_NODES * HDIM * 4;
static constexpr size_t OFF_AGG    = OFF_HNEW   + (size_t)N_NODES * HDIM * 4;
static constexpr size_t OFF_HBF    = OFF_AGG    + (size_t)N_NODES * HDIM * 4;
static constexpr size_t OFF_DEG    = OFF_HBF    + (size_t)N_NODES * HDIM * 2;
static constexpr size_t OFF_BCNT   = OFF_DEG    + (size_t)N_NODES * 4;
static constexpr size_t OFF_BSUM   = OFF_BCNT   + 256;
static constexpr size_t OFF_BSUMSQ = OFF_BSUM   + (size_t)NB_ * HDIM * 4;
static constexpr size_t OFF_MEAN   = OFF_BSUMSQ + (size_t)NB_ * HDIM * 4;
static constexpr size_t OFF_RSTD   = OFF_MEAN   + (size_t)NB_ * HDIM * 4;
static constexpr size_t OFF_EMBW1F = OFF_RSTD   + (size_t)NB_ * HDIM * 4;
static constexpr size_t OFF_EMBW2F = OFF_EMBW1F + (size_t)EMB_KPAD * 128 * 2;
static constexpr size_t OFF_MSGW1F = OFF_EMBW2F + (size_t)128 * 128 * 2;
static constexpr size_t OFF_MSGW2F = OFF_MSGW1F + (size_t)NLAY * MSG_KPAD * 128 * 2;
static constexpr size_t OFF_UPDW1F = OFF_MSGW2F + (size_t)NLAY * 128 * 128 * 2;
static constexpr size_t OFF_UPDW2F = OFF_UPDW1F + (size_t)NLAY * UPD_KPAD * 128 * 2;

static constexpr int GEMM_LDS_BYTES =
    (9 * 8 * 32 * 16 + 4 * 8 * 32 * 16 + 8 * 16 * MSG_KPAD) * 2 + 8 * 16 * 4 +
    8 * 16 * HDIM * 2;  // 213504
static constexpr int EMB_LDS_BYTES =
    (1 * 8 * 32 * 16 + 4 * 8 * 32 * 16 + 8 * 16 * EMB_KPAD + 8 * 16 * HDIM) * 2;  // 81920

extern "C" void kernel_launch(void* const* d_in, const int* in_sizes, int n_in,
                              void* d_out, int out_size, void* d_ws, size_t ws_size,
                              hipStream_t stream) {
  const float* u      = (const float*)d_in[0];
  const float* pos    = (const float*)d_in[1];
  const int*   ei     = (const int*)d_in[2];
  const int*   batch  = (const int*)d_in[3];
  const float* embW1  = (const float*)d_in[4];
  const float* emb_b1 = (const float*)d_in[5];
  const float* embW2  = (const float*)d_in[6];
  const float* emb_b2 = (const float*)d_in[7];
  const float* msgW1  = (const float*)d_in[8];
  const float* msg_b1 = (const float*)d_in[9];
  const float* msgW2  = (const float*)d_in[10];
  const float* msg_b2 = (const float*)d_in[11];
  const float* updW1  = (const float*)d_in[12];
  const float* upd_b1 = (const float*)d_in[13];
  const float* updW2  = (const float*)d_in[14];
  const float* upd_b2 = (const float*)d_in[15];
  const float* c1W = (const float*)d_in[16];
  const float* c1b = (const float*)d_in[17];
  const float* c2W = (const float*)d_in[18];
  const float* c2b = (const float*)d_in[19];
  const float* c3W = (const float*)d_in[20];
  const float* c3b = (const float*)d_in[21];
  float* out = (float*)d_out;

  char* ws = (char*)d_ws;
  float* h      = (float*)(ws + OFF_H);
  float* hnew   = (float*)(ws + OFF_HNEW);
  float* agg    = (float*)(ws + OFF_AGG);
  unsigned short* hbf = (unsigned short*)(ws + OFF_HBF);
  float* deg    = (float*)(ws + OFF_DEG);
  float* bcnt   = (float*)(ws + OFF_BCNT);
  float* bsum   = (float*)(ws + OFF_BSUM);
  float* bsumsq = (float*)(ws + OFF_BSUMSQ);
  float* meanb  = (float*)(ws + OFF_MEAN);
  float* rstdb  = (float*)(ws + OFF_RSTD);
  unsigned short* embW1f = (unsigned short*)(ws + OFF_EMBW1F);
  unsigned short* embW2f = (unsigned short*)(ws + OFF_EMBW2F);
  unsigned short* msgW1f = (unsigned short*)(ws + OFF_MSGW1F);
  unsigned short* msgW2f = (unsigned short*)(ws + OFF_MSGW2F);
  unsigned short* updW1f = (unsigned short*)(ws + OFF_UPDW1F);
  unsigned short* updW2f = (unsigned short*)(ws + OFF_UPDW2F);

  (void)hipFuncSetAttribute(reinterpret_cast<const void*>(&message_kernel),
                            hipFuncAttributeMaxDynamicSharedMemorySize, GEMM_LDS_BYTES);
  (void)hipFuncSetAttribute(reinterpret_cast<const void*>(&update_kernel),
                            hipFuncAttributeMaxDynamicSharedMemorySize, GEMM_LDS_BYTES);
  (void)hipFuncSetAttribute(reinterpret_cast<const void*>(&embed_kernel),
                            hipFuncAttributeMaxDynamicSharedMemorySize, EMB_LDS_BYTES);

  // degree + batch counts (invariant across layers)
  zero_kernel<<<(N_NODES + 255) / 256, 256, 0, stream>>>(deg, N_NODES);
  zero_kernel<<<1, 256, 0, stream>>>(bcnt, NB_);
  count_deg_kernel<<<N_EDGES / 256, 256, 0, stream>>>(ei, deg);
  count_batch_kernel<<<N_NODES / 256, 256, 0, stream>>>(batch, bcnt);

  // repack weights to bf16 WMMA fragment order
  pack_weights_kernel<<<(EMB_KPAD * 128 + 255) / 256, 256, 0, stream>>>(embW1, 28, EMB_KPAD, embW1f);
  pack_weights_kernel<<<(128 * 128 + 255) / 256, 256, 0, stream>>>(embW2, 128, 128, embW2f);
  for (int l = 0; l < NLAY; ++l) {
    pack_weights_kernel<<<(MSG_KPAD * 128 + 255) / 256, 256, 0, stream>>>(
        msgW1 + (size_t)l * 284 * 128, 284, MSG_KPAD, msgW1f + (size_t)l * MSG_KPAD * 128);
    pack_weights_kernel<<<(128 * 128 + 255) / 256, 256, 0, stream>>>(
        msgW2 + (size_t)l * 128 * 128, 128, 128, msgW2f + (size_t)l * 128 * 128);
    pack_weights_kernel<<<(UPD_KPAD * 128 + 255) / 256, 256, 0, stream>>>(
        updW1 + (size_t)l * 257 * 128, 257, UPD_KPAD, updW1f + (size_t)l * UPD_KPAD * 128);
    pack_weights_kernel<<<(128 * 128 + 255) / 256, 256, 0, stream>>>(
        updW2 + (size_t)l * 128 * 128, 128, 128, updW2f + (size_t)l * 128 * 128);
  }

  embed_kernel<<<N_NODES / 128, 256, EMB_LDS_BYTES, stream>>>(u, pos, embW1f, emb_b1,
                                                              embW2f, emb_b2, h, hbf);

  for (int l = 0; l < NLAY; ++l) {
    zero_kernel<<<(N_NODES * HDIM + 255) / 256, 256, 0, stream>>>(agg, N_NODES * HDIM);
    zero_kernel<<<(2 * NB_ * HDIM + 255) / 256, 256, 0, stream>>>(bsum, 2 * NB_ * HDIM);
    message_kernel<<<N_EDGES / 128, 256, GEMM_LDS_BYTES, stream>>>(
        hbf, u, pos, ei, msgW1f + (size_t)l * MSG_KPAD * 128, msg_b1 + l * HDIM,
        msgW2f + (size_t)l * 128 * 128, msg_b2 + l * HDIM, agg);
    update_kernel<<<N_NODES / 128, 256, GEMM_LDS_BYTES, stream>>>(
        hbf, h, agg, deg, pos, batch, updW1f + (size_t)l * UPD_KPAD * 128, upd_b1 + l * HDIM,
        updW2f + (size_t)l * 128 * 128, upd_b2 + l * HDIM, hnew, bsum, bsumsq);
    stats_kernel<<<(NB_ * HDIM + 255) / 256, 256, 0, stream>>>(bsum, bsumsq, bcnt, meanb, rstdb);
    apply_norm_kernel<<<(N_NODES * HDIM + 255) / 256, 256, 0, stream>>>(hnew, batch, meanb,
                                                                        rstdb, h, hbf);
  }

  head_kernel<<<N_NODES / 8, 256, 0, stream>>>(h, u, c1W, c1b, c2W, c2b, c3W, c3b, out);

  (void)in_sizes; (void)n_in; (void)out_size; (void)ws_size;
}
